// NystromAttention_44719199485976
// MI455X (gfx1250) — matrix operand order repaired and checked by reading, not compiled, not run on hardware
//
#include <hip/hip_runtime.h>
#include <hip/hip_bf16.h>
#include <math.h>

// ---------------------------------------------------------------------------
// Types
// ---------------------------------------------------------------------------
typedef __bf16 bf16_t;
typedef __attribute__((ext_vector_type(16))) __bf16 v16bf;
typedef __attribute__((ext_vector_type(8)))  __bf16 v8bf;
typedef __attribute__((ext_vector_type(2)))  __bf16 v2bf;
typedef __attribute__((ext_vector_type(8)))  float  v8f;
typedef __attribute__((ext_vector_type(2)))  float  v2f;

#define B_   4
#define N_   4096
#define C_   512
#define H_   8
#define HD_  64
#define M_   64
#define BH_  (B_*H_)          // 32
#define ROWS_ (B_*N_)         // 16384
#define QK_SCALE 0.3535533905932738f   // 64^-0.25

// ---------------------------------------------------------------------------
// bf16 helpers: native converts (v_cvt_pk_bf16_f32 on gfx1250)
// ---------------------------------------------------------------------------
__device__ __forceinline__ bf16_t f2bf(float f) { return (bf16_t)f; }
__device__ __forceinline__ float  bf2f(bf16_t b) { return (float)b; }
__device__ __forceinline__ unsigned pack_bf2(float lo, float hi) {
  v2bf pk; pk[0] = (bf16_t)lo; pk[1] = (bf16_t)hi;
  return __builtin_bit_cast(unsigned, pk);
}
__device__ __forceinline__ uint2 pack_bf4(float4 v) {
  uint2 pk; pk.x = pack_bf2(v.x, v.y); pk.y = pack_bf2(v.z, v.w); return pk;
}
__device__ __forceinline__ v8f v8f_zero() {
  v8f z = {0.f,0.f,0.f,0.f,0.f,0.f,0.f,0.f}; return z;
}

// ---------------------------------------------------------------------------
// WMMA fragment builders, using this code's contiguous k order, not the ISA table order (wave32).
// A (16x32 bf16): lane<16 -> row M=lane, halves K0..15
//                 lane>=16 -> row M=lane-16, halves K16..31
// B (32x16 bf16): lane<16 -> col N=lane, halves K0..15
//                 lane>=16 -> col N=lane-16, halves K16..31
// C/D (16x16 f32): vgpr i, lane<16 -> (M=i,   N=lane)
//                          lane>=16 -> (M=i+8, N=lane-16)
// ---------------------------------------------------------------------------
__device__ __forceinline__ v16bf load_a_frag(const bf16_t* A, int lda) {
  int lane = threadIdx.x & 31;
  int row  = lane & 15;
  int koff = (lane < 16) ? 0 : 16;
  const bf16_t* p = A + row * lda + koff;
  v8bf lo = *(const v8bf*)(p);
  v8bf hi = *(const v8bf*)(p + 8);
  return __builtin_shufflevector(lo, hi, 0,1,2,3,4,5,6,7,8,9,10,11,12,13,14,15);
}

// B = W^T where W is row-major [n][k]  (computes A @ W^T) -> one 32B load
__device__ __forceinline__ v16bf load_b_fragT(const bf16_t* W, int ldw) {
  int lane = threadIdx.x & 31;
  int n    = lane & 15;
  int koff = (lane < 16) ? 0 : 16;
  return *(const v16bf*)(W + n * ldw + koff);
}

__device__ __forceinline__ v8f wmma_bf16(v16bf a, v16bf b, v8f c) {
  return __builtin_amdgcn_wmma_f32_16x16x32_bf16(false, a, false, b,
                                                 (short)0, c, false, false);
}

// fp32 WMMA 16x16x4 fragments
__device__ __forceinline__ v2f load_a2(const float* A, int lda) {
  int lane = threadIdx.x & 31;
  int row  = lane & 15;
  int k    = (lane < 16) ? 0 : 2;
  v2f a; a[0] = A[row*lda + k]; a[1] = A[row*lda + k + 1]; return a;
}
__device__ __forceinline__ v2f load_b2(const float* Bm, int ldb) {
  int lane = threadIdx.x & 31;
  int n    = lane & 15;
  int k    = (lane < 16) ? 0 : 2;
  v2f b; b[0] = Bm[k*ldb + n]; b[1] = Bm[(k+1)*ldb + n]; return b;
}

// ---------------------------------------------------------------------------
// 0) init: zero the global column-sum-max slot
// ---------------------------------------------------------------------------
__global__ void k_init(int* colmax) {
  if (blockIdx.x == 0 && threadIdx.x == 0) colmax[0] = 0;  // bits of 0.0f
}

// ---------------------------------------------------------------------------
// 1) QKV GEMM: [16384,512] @ [1536,512]^T, scale folded into Q,K.
//    Block = 128 thr (4 waves), tile 64x64, K-loop 512/32.
//    Staging: depth-4 batches (2 A + 2 W chunks) -> clause loads, 1 drain.
// ---------------------------------------------------------------------------
__global__ void k_qkv(const float* __restrict__ x, const float* __restrict__ w,
                      bf16_t* __restrict__ Qb, bf16_t* __restrict__ Kb,
                      bf16_t* __restrict__ Vb) {
  __shared__ __align__(32) bf16_t As[64*32];
  __shared__ __align__(32) bf16_t Ws[64*32];
  int m0 = blockIdx.x * 64;
  int n0 = blockIdx.y * 64;
  int tid = threadIdx.x, wid = tid >> 5, lane = tid & 31;
  v8f acc[4] = {v8f_zero(), v8f_zero(), v8f_zero(), v8f_zero()};
  for (int kt = 0; kt < 16; ++kt) {
    int k0 = kt * 32;
#pragma unroll
    for (int hb = 0; hb < 2; ++hb) {      // depth-4 staging batches
      int e0 = (tid + (2*hb+0)*128) * 4;
      int e1 = (tid + (2*hb+1)*128) * 4;
      int r0 = e0 >> 5, c0 = e0 & 31;
      int r1 = e1 >> 5, c1 = e1 & 31;
      const float* ga0 = x + (size_t)(m0 + r0)*C_ + k0 + c0;
      const float* ga1 = x + (size_t)(m0 + r1)*C_ + k0 + c1;
      const float* gw0 = w + (size_t)(n0 + r0)*C_ + k0 + c0;
      const float* gw1 = w + (size_t)(n0 + r1)*C_ + k0 + c1;
      float4 a0 = *(const float4*)ga0;
      float4 a1 = *(const float4*)ga1;
      float4 w0 = *(const float4*)gw0;
      float4 w1 = *(const float4*)gw1;
      if (kt < 15) {
        __builtin_prefetch(ga0 + 32, 0, 1); __builtin_prefetch(ga1 + 32, 0, 1);
        __builtin_prefetch(gw0 + 32, 0, 1); __builtin_prefetch(gw1 + 32, 0, 1);
      }
      *(uint2*)&As[e0] = pack_bf4(a0);
      *(uint2*)&As[e1] = pack_bf4(a1);
      *(uint2*)&Ws[e0] = pack_bf4(w0);
      *(uint2*)&Ws[e1] = pack_bf4(w1);
    }
    __syncthreads();
    v16bf a = load_a_frag(As + wid*16*32, 32);
#pragma unroll
    for (int j = 0; j < 4; ++j) {
      v16bf b = load_b_fragT(Ws + j*16*32, 32);
      acc[j] = wmma_bf16(a, b, acc[j]);
    }
    __syncthreads();
  }
  const float scale = QK_SCALE;
#pragma unroll
  for (int j = 0; j < 4; ++j)
#pragma unroll
    for (int i = 0; i < 8; ++i) {
      int m  = m0 + wid*16 + ((lane < 16) ? i : i + 8);
      int gc = n0 + j*16 + (lane & 15);
      float v = acc[j][i];
      int which = gc >> 9;        // 0=Q 1=K 2=V
      int cc = gc & 511;
      int h = cc >> 6, d = cc & 63;
      int bi = m >> 12, ni = m & (N_-1);
      size_t dst = (((size_t)bi*H_ + h)*N_ + ni)*HD_ + d;
      if (which == 0)      Qb[dst] = f2bf(v * scale);
      else if (which == 1) Kb[dst] = f2bf(v * scale);
      else                 Vb[dst] = f2bf(v);
    }
}

// ---------------------------------------------------------------------------
// 2) landmark mean-pool: Ql/Kl[bh][m][d] = mean over 64 consecutive n
// ---------------------------------------------------------------------------
__global__ void k_pool(const bf16_t* __restrict__ Qb, const bf16_t* __restrict__ Kb,
                       bf16_t* __restrict__ Ql, bf16_t* __restrict__ Kl) {
  int g = blockIdx.x * blockDim.x + threadIdx.x;
  if (g >= BH_*M_*HD_) return;
  int bh = g >> 12, m = (g >> 6) & 63, d = g & 63;
  size_t base = ((size_t)bh*N_ + m*64)*HD_ + d;
  float sq = 0.f, sk = 0.f;
#pragma unroll 4
  for (int j = 0; j < 64; ++j) {
    sq += bf2f(Qb[base + (size_t)j*HD_]);
    sk += bf2f(Kb[base + (size_t)j*HD_]);
  }
  Ql[g] = f2bf(sq * (1.0f/64.0f));
  Kl[g] = f2bf(sk * (1.0f/64.0f));
}

// ---------------------------------------------------------------------------
// 3) kernel_2 = softmax(Ql @ Kl^T) + global atomic max of column sums
// ---------------------------------------------------------------------------
__global__ void k_kernel2(const bf16_t* __restrict__ Ql, const bf16_t* __restrict__ Kl,
                          float* __restrict__ k2, int* __restrict__ colmax) {
  int bh = blockIdx.x;
  const bf16_t* A = Ql + (size_t)bh*M_*HD_;
  const bf16_t* W = Kl + (size_t)bh*M_*HD_;
  __shared__ float S[64*64];
  int tid = threadIdx.x, wid = tid >> 5, lane = tid & 31;
  v8f acc[4] = {v8f_zero(), v8f_zero(), v8f_zero(), v8f_zero()};
  for (int kt = 0; kt < 2; ++kt) {
    v16bf a = load_a_frag(A + wid*16*HD_ + kt*32, HD_);
#pragma unroll
    for (int j = 0; j < 4; ++j) {
      v16bf b = load_b_fragT(W + j*16*HD_ + kt*32, HD_);
      acc[j] = wmma_bf16(a, b, acc[j]);
    }
  }
#pragma unroll
  for (int j = 0; j < 4; ++j)
#pragma unroll
    for (int i = 0; i < 8; ++i) {
      int m = wid*16 + ((lane < 16) ? i : i + 8);
      int c = j*16 + (lane & 15);
      S[m*64 + c] = acc[j][i];
    }
  __syncthreads();
  if (tid < 64) {
    float mx = -1e30f;
    for (int c = 0; c < 64; ++c) mx = fmaxf(mx, S[tid*64 + c]);
    float sum = 0.f;
    for (int c = 0; c < 64; ++c) { float e = expf(S[tid*64 + c] - mx); S[tid*64 + c] = e; sum += e; }
    float inv = 1.0f / sum;
    for (int c = 0; c < 64; ++c) {
      float v = S[tid*64 + c] * inv;
      S[tid*64 + c] = v;
      k2[((size_t)bh*M_ + tid)*M_ + c] = v;
    }
  }
  __syncthreads();
  if (tid < 64) {   // column sums of the softmax'd matrix; row sums are 1
    float cs = 0.f;
    for (int r = 0; r < 64; ++r) cs += S[r*64 + tid];
    atomicMax(colmax, __float_as_int(cs));   // non-negative -> int order == float order
  }
}

// ---------------------------------------------------------------------------
// 4) kernel_3 logits: [bh] Ql(64x64) @ K^T(64 x 4096) -> f32 [bh][64][4096]
// ---------------------------------------------------------------------------
__global__ void k_logits(const bf16_t* __restrict__ Ql, const bf16_t* __restrict__ Kb,
                         float* __restrict__ logits) {
  int nt = blockIdx.x, bh = blockIdx.y;
  int n0 = nt * 64;
  const bf16_t* A = Ql + (size_t)bh*M_*HD_;
  const bf16_t* W = Kb + ((size_t)bh*N_ + n0)*HD_;
  int tid = threadIdx.x, wid = tid >> 5, lane = tid & 31;
  v8f acc[4] = {v8f_zero(), v8f_zero(), v8f_zero(), v8f_zero()};
  for (int kt = 0; kt < 2; ++kt) {
    v16bf a = load_a_frag(A + wid*16*HD_ + kt*32, HD_);
#pragma unroll
    for (int j = 0; j < 4; ++j) {
      v16bf b = load_b_fragT(W + j*16*HD_ + kt*32, HD_);
      acc[j] = wmma_bf16(a, b, acc[j]);
    }
  }
#pragma unroll
  for (int j = 0; j < 4; ++j)
#pragma unroll
    for (int i = 0; i < 8; ++i) {
      int m = wid*16 + ((lane < 16) ? i : i + 8);
      int c = j*16 + (lane & 15);
      logits[((size_t)bh*M_ + m)*N_ + n0 + c] = acc[j][i];
    }
}

// ---------------------------------------------------------------------------
// 5) row softmax over N=4096, in place (one block per row)
// ---------------------------------------------------------------------------
__global__ void k_softmax_rows(float* __restrict__ logits) {
  float* p = logits + (size_t)blockIdx.x * N_;
  __shared__ float red[256];
  int t = threadIdx.x;
  float m = -1e30f;
  for (int i = t; i < N_; i += 256) m = fmaxf(m, p[i]);
  red[t] = m; __syncthreads();
  for (int s = 128; s > 0; s >>= 1) { if (t < s) red[t] = fmaxf(red[t], red[t+s]); __syncthreads(); }
  float rmax = red[0]; __syncthreads();
  float sum = 0.f;
  for (int i = t; i < N_; i += 256) { float e = expf(p[i] - rmax); p[i] = e; sum += e; }
  red[t] = sum; __syncthreads();
  for (int s = 128; s > 0; s >>= 1) { if (t < s) red[t] += red[t+s]; __syncthreads(); }
  float inv = 1.0f / red[0];
  for (int i = t; i < N_; i += 256) p[i] *= inv;
}

// ---------------------------------------------------------------------------
// 6) K3V[bh] = P(64x4096) @ V(4096x64), chunked K=32 through LDS.
//    Depth-4 staging batches; V chunk stored transposed in LDS.
// ---------------------------------------------------------------------------
__global__ void k_k3v(const float* __restrict__ P, const bf16_t* __restrict__ Vb,
                      float* __restrict__ K3V) {
  int bh = blockIdx.x;
  __shared__ __align__(32) bf16_t Pch[64*32];   // [m][k]
  __shared__ __align__(32) bf16_t Vt[64*32];    // [d][k]  (transposed V chunk)
  int tid = threadIdx.x, wid = tid >> 5, lane = tid & 31;
  v8f acc[4] = {v8f_zero(), v8f_zero(), v8f_zero(), v8f_zero()};
  for (int ch = 0; ch < N_/32; ++ch) {
    int n0 = ch * 32;
#pragma unroll
    for (int hb = 0; hb < 2; ++hb) {
      int e0 = (tid + (2*hb+0)*128) * 4;
      int e1 = (tid + (2*hb+1)*128) * 4;
      int r0 = e0 >> 5, c0 = e0 & 31;
      int r1 = e1 >> 5, c1 = e1 & 31;
      int k0_ = e0 >> 6, d0 = e0 & 63;
      int k1_ = e1 >> 6, d1 = e1 & 63;
      float4 p0 = *(const float4*)(P + ((size_t)bh*M_ + r0)*N_ + n0 + c0);
      float4 p1 = *(const float4*)(P + ((size_t)bh*M_ + r1)*N_ + n0 + c1);
      uint2  q0 = *(const uint2*)(Vb + ((size_t)bh*N_ + n0 + k0_)*HD_ + d0);
      uint2  q1 = *(const uint2*)(Vb + ((size_t)bh*N_ + n0 + k1_)*HD_ + d1);
      *(uint2*)&Pch[e0] = pack_bf4(p0);
      *(uint2*)&Pch[e1] = pack_bf4(p1);
      bf16_t t0[4]; *(uint2*)t0 = q0;
      bf16_t t1[4]; *(uint2*)t1 = q1;
#pragma unroll
      for (int i = 0; i < 4; ++i) {
        Vt[(d0+i)*32 + k0_] = t0[i];
        Vt[(d1+i)*32 + k1_] = t1[i];
      }
    }
    __syncthreads();
    v16bf a = load_a_frag(Pch + wid*16*32, 32);
#pragma unroll
    for (int j = 0; j < 4; ++j) {
      v16bf b = load_b_fragT(Vt + j*16*32, 32);   // row n=d, cols k
      acc[j] = wmma_bf16(a, b, acc[j]);
    }
    __syncthreads();
  }
#pragma unroll
  for (int j = 0; j < 4; ++j)
#pragma unroll
    for (int i = 0; i < 8; ++i) {
      int m = wid*16 + ((lane < 16) ? i : i + 8);
      int c = j*16 + (lane & 15);
      K3V[((size_t)bh*M_ + m)*M_ + c] = acc[j][i];
    }
}

// ---------------------------------------------------------------------------
// 7) Newton-Schulz pieces (fp32 WMMA): C = alpha*I + beta*(A@B), batched bh
// ---------------------------------------------------------------------------
__global__ void k_mm64(const float* __restrict__ Ab, const float* __restrict__ Bb,
                       float* __restrict__ Cb, float alpha, float beta) {
  int bh = blockIdx.x;
  const float* A  = Ab + (size_t)bh*M_*M_;
  const float* Bm = Bb + (size_t)bh*M_*M_;
  float*       Cm = Cb + (size_t)bh*M_*M_;
  int tid = threadIdx.x, wid = tid >> 5, lane = tid & 31;
  v8f acc[4] = {v8f_zero(), v8f_zero(), v8f_zero(), v8f_zero()};
  for (int kt = 0; kt < 16; ++kt) {
    v2f a = load_a2(A + wid*16*M_ + kt*4, M_);
#pragma unroll
    for (int j = 0; j < 4; ++j) {
      v2f b = load_b2(Bm + kt*4*M_ + j*16, M_);
      acc[j] = __builtin_amdgcn_wmma_f32_16x16x4_f32(false, a, false, b,
                                                     (short)0, acc[j], false, false);
    }
  }
#pragma unroll
  for (int j = 0; j < 4; ++j)
#pragma unroll
    for (int i = 0; i < 8; ++i) {
      int m = wid*16 + ((lane < 16) ? i : i + 8);
      int c = j*16 + (lane & 15);
      Cm[m*M_ + c] = beta*acc[j][i] + ((m == c) ? alpha : 0.0f);
    }
}

// W2T[bh] = (A@B)^T stored bf16  (A=Vinv, B=K3V)
__global__ void k_mm64_bf16T(const float* __restrict__ Ab, const float* __restrict__ Bb,
                             bf16_t* __restrict__ Ct) {
  int bh = blockIdx.x;
  const float* A  = Ab + (size_t)bh*M_*M_;
  const float* Bm = Bb + (size_t)bh*M_*M_;
  bf16_t*      Cm = Ct + (size_t)bh*M_*M_;
  int tid = threadIdx.x, wid = tid >> 5, lane = tid & 31;
  v8f acc[4] = {v8f_zero(), v8f_zero(), v8f_zero(), v8f_zero()};
  for (int kt = 0; kt < 16; ++kt) {
    v2f a = load_a2(A + wid*16*M_ + kt*4, M_);
#pragma unroll
    for (int j = 0; j < 4; ++j) {
      v2f b = load_b2(Bm + kt*4*M_ + j*16, M_);
      acc[j] = __builtin_amdgcn_wmma_f32_16x16x4_f32(false, a, false, b,
                                                     (short)0, acc[j], false, false);
    }
  }
#pragma unroll
  for (int j = 0; j < 4; ++j)
#pragma unroll
    for (int i = 0; i < 8; ++i) {
      int m = wid*16 + ((lane < 16) ? i : i + 8);
      int c = j*16 + (lane & 15);
      Cm[c*M_ + m] = f2bf(acc[j][i]);      // transposed store
    }
}

// dst = alpha*I + beta*src (elementwise over [bh][64][64])
__global__ void k_axpyI(const float* __restrict__ src, float* __restrict__ dst,
                        float alpha, float beta) {
  int g = blockIdx.x * blockDim.x + threadIdx.x;
  if (g >= BH_*M_*M_) return;
  int rc = g & 4095; int r = rc >> 6, c = rc & 63;
  dst[g] = beta*src[g] + ((r == c) ? alpha : 0.0f);
}

// V0 = K^T / (max_colsum * max_rowsum);  rowsums of softmax rows == 1
__global__ void k_ns_init(const float* __restrict__ k2, float* __restrict__ V0,
                          const float* __restrict__ colmax) {
  int g = blockIdx.x * blockDim.x + threadIdx.x;
  if (g >= BH_*M_*M_) return;
  int bh = g >> 12, r = (g >> 6) & 63, c = g & 63;
  float inv = 1.0f / colmax[0];
  V0[g] = k2[(size_t)bh*M_*M_ + c*M_ + r] * inv;
}

// ---------------------------------------------------------------------------
// 8) attention apply: out_head = softmax(Q @ Kl^T) @ W2  -> attn[B,N,C] bf16
// ---------------------------------------------------------------------------
__global__ void k_attn(const bf16_t* __restrict__ Qb, const bf16_t* __restrict__ Kl,
                       const bf16_t* __restrict__ W2T, bf16_t* __restrict__ attn) {
  int nt = blockIdx.x, bh = blockIdx.y;
  int n0 = nt * 64;
  const bf16_t* Qbh = Qb  + ((size_t)bh*N_ + n0)*HD_;
  const bf16_t* Klb = Kl  + (size_t)bh*M_*HD_;
  const bf16_t* Wbh = W2T + (size_t)bh*M_*M_;
  __shared__ float  S[64*64];
  __shared__ __align__(32) bf16_t Pm[64*64];
  int tid = threadIdx.x, wid = tid >> 5, lane = tid & 31;
  {
    v8f acc[4] = {v8f_zero(), v8f_zero(), v8f_zero(), v8f_zero()};
    for (int kt = 0; kt < 2; ++kt) {
      v16bf a = load_a_frag(Qbh + wid*16*HD_ + kt*32, HD_);
#pragma unroll
      for (int j = 0; j < 4; ++j) {
        v16bf b = load_b_fragT(Klb + j*16*HD_ + kt*32, HD_);
        acc[j] = wmma_bf16(a, b, acc[j]);
      }
    }
#pragma unroll
    for (int j = 0; j < 4; ++j)
#pragma unroll
      for (int i = 0; i < 8; ++i) {
        int m = wid*16 + ((lane < 16) ? i : i + 8);
        int c = j*16 + (lane & 15);
        S[m*64 + c] = acc[j][i];
      }
  }
  __syncthreads();
  if (tid < 64) {
    float mx = -1e30f;
    for (int c = 0; c < 64; ++c) mx = fmaxf(mx, S[tid*64 + c]);
    float sum = 0.f;
    for (int c = 0; c < 64; ++c) { float e = expf(S[tid*64 + c] - mx); S[tid*64 + c] = e; sum += e; }
    float inv = 1.0f / sum;
    for (int c = 0; c < 64; ++c) Pm[tid*64 + c] = f2bf(S[tid*64 + c] * inv);
  }
  __syncthreads();
  v8f acc[4] = {v8f_zero(), v8f_zero(), v8f_zero(), v8f_zero()};
  for (int kt = 0; kt < 2; ++kt) {
    v16bf a = load_a_frag(Pm + wid*16*64 + kt*32, 64);
#pragma unroll
    for (int j = 0; j < 4; ++j) {
      v16bf b = load_b_fragT(Wbh + j*16*M_ + kt*32, M_);  // W2T rows n, cols k -> A@W2
      acc[j] = wmma_bf16(a, b, acc[j]);
    }
  }
  int b_ = bh >> 3, h = bh & 7;
#pragma unroll
  for (int j = 0; j < 4; ++j)
#pragma unroll
    for (int i = 0; i < 8; ++i) {
      int m = wid*16 + ((lane < 16) ? i : i + 8);
      int c = j*16 + (lane & 15);
      attn[((size_t)(b_*N_ + n0 + m))*C_ + h*HD_ + c] = f2bf(acc[j][i]);
    }
}

// ---------------------------------------------------------------------------
// 9) output projection: out = attn @ proj_w^T + proj_b (f32 out), tile 64x64
// ---------------------------------------------------------------------------
__global__ void k_proj(const bf16_t* __restrict__ attn, const float* __restrict__ w,
                       const float* __restrict__ bias, float* __restrict__ out) {
  __shared__ __align__(32) bf16_t As[64*32];
  __shared__ __align__(32) bf16_t Ws[64*32];
  int m0 = blockIdx.x * 64;
  int n0 = blockIdx.y * 64;
  int tid = threadIdx.x, wid = tid >> 5, lane = tid & 31;
  v8f acc[4] = {v8f_zero(), v8f_zero(), v8f_zero(), v8f_zero()};
  for (int kt = 0; kt < 16; ++kt) {
    int k0 = kt * 32;
#pragma unroll
    for (int hb = 0; hb < 2; ++hb) {      // depth-4 staging batches
      int e0 = (tid + (2*hb+0)*128) * 4;
      int e1 = (tid + (2*hb+1)*128) * 4;
      int r0 = e0 >> 5, c0 = e0 & 31;
      int r1 = e1 >> 5, c1 = e1 & 31;
      const bf16_t* ga0 = attn + (size_t)(m0 + r0)*C_ + k0 + c0;
      const bf16_t* ga1 = attn + (size_t)(m0 + r1)*C_ + k0 + c1;
      const float*  gw0 = w    + (size_t)(n0 + r0)*C_ + k0 + c0;
      const float*  gw1 = w    + (size_t)(n0 + r1)*C_ + k0 + c1;
      uint2  a0 = *(const uint2*)ga0;
      uint2  a1 = *(const uint2*)ga1;
      float4 w0 = *(const float4*)gw0;
      float4 w1 = *(const float4*)gw1;
      if (kt < 15) {
        __builtin_prefetch(ga0 + 32, 0, 1); __builtin_prefetch(ga1 + 32, 0, 1);
        __builtin_prefetch(gw0 + 32, 0, 1); __builtin_prefetch(gw1 + 32, 0, 1);
      }
      *(uint2*)&As[e0] = a0;
      *(uint2*)&As[e1] = a1;
      *(uint2*)&Ws[e0] = pack_bf4(w0);
      *(uint2*)&Ws[e1] = pack_bf4(w1);
    }
    __syncthreads();
    v16bf a = load_a_frag(As + wid*16*32, 32);
#pragma unroll
    for (int j = 0; j < 4; ++j) {
      v16bf b = load_b_fragT(Ws + j*16*32, 32);
      acc[j] = wmma_bf16(a, b, acc[j]);
    }
    __syncthreads();
  }
#pragma unroll
  for (int j = 0; j < 4; ++j)
#pragma unroll
    for (int i = 0; i < 8; ++i) {
      int m  = m0 + wid*16 + ((lane < 16) ? i : i + 8);
      int gc = n0 + j*16 + (lane & 15);
      out[(size_t)m*C_ + gc] = acc[j][i] + bias[gc];
    }
}

// ---------------------------------------------------------------------------
// launcher
// ---------------------------------------------------------------------------
extern "C" void kernel_launch(void* const* d_in, const int* in_sizes, int n_in,
                              void* d_out, int out_size, void* d_ws, size_t ws_size,
                              hipStream_t stream) {
  const float* x      = (const float*)d_in[0];
  const float* qkv_w  = (const float*)d_in[1];
  const float* proj_w = (const float*)d_in[2];
  const float* proj_b = (const float*)d_in[3];
  float* out = (float*)d_out;

  char* ws = (char*)d_ws;
  size_t off = 0;
  auto alloc = [&](size_t bytes) -> void* {
    void* p = ws + off;
    off = (off + bytes + 255) & ~(size_t)255;
    return p;
  };
  const size_t nBHND = (size_t)BH_ * N_ * HD_;   // 8,388,608
  const size_t nBH64 = (size_t)BH_ * M_ * M_;    // 131,072

  bf16_t* Qb     = (bf16_t*)alloc(nBHND * 2);
  bf16_t* Kb     = (bf16_t*)alloc(nBHND * 2);
  bf16_t* Vb     = (bf16_t*)alloc(nBHND * 2);
  bf16_t* Ql     = (bf16_t*)alloc(nBH64 * 2);
  bf16_t* Kl     = (bf16_t*)alloc(nBH64 * 2);
  float*  k2     = (float*) alloc(nBH64 * 4);
  float*  logits = (float*) alloc(nBHND * 4);
  float*  K3V    = (float*) alloc(nBH64 * 4);
  float*  V0     = (float*) alloc(nBH64 * 4);
  float*  V1     = (float*) alloc(nBH64 * 4);
  float*  KV     = (float*) alloc(nBH64 * 4);
  float*  U1     = (float*) alloc(nBH64 * 4);
  float*  U2     = (float*) alloc(nBH64 * 4);
  bf16_t* W2T    = (bf16_t*)alloc(nBH64 * 2);
  bf16_t* attn   = (bf16_t*)alloc((size_t)ROWS_ * C_ * 2);
  int*    colmax = (int*)   alloc(256);

  k_init<<<1, 32, 0, stream>>>(colmax);
  k_qkv<<<dim3(ROWS_/64, (3*C_)/64), 128, 0, stream>>>(x, qkv_w, Qb, Kb, Vb);
  k_pool<<<(BH_*M_*HD_ + 255)/256, 256, 0, stream>>>(Qb, Kb, Ql, Kl);
  k_kernel2<<<BH_, 128, 0, stream>>>(Ql, Kl, k2, colmax);
  k_logits<<<dim3(N_/64, BH_), 128, 0, stream>>>(Ql, Kb, logits);
  k_softmax_rows<<<BH_*M_, 256, 0, stream>>>(logits);
  k_k3v<<<BH_, 128, 0, stream>>>(logits, Vb, K3V);
  k_ns_init<<<(BH_*M_*M_ + 255)/256, 256, 0, stream>>>(k2, V0, (const float*)colmax);

  float* Vcur = V0; float* Vnxt = V1;
  for (int it = 0; it < 6; ++it) {
    k_mm64<<<BH_, 128, 0, stream>>>(k2, Vcur, KV, 0.0f, 1.0f);                 // KV = K@V
    k_axpyI<<<(BH_*M_*M_ + 255)/256, 256, 0, stream>>>(KV, U1, 7.0f, -1.0f);   // 7I - KV
    k_mm64<<<BH_, 128, 0, stream>>>(KV, U1, U2, 15.0f, -1.0f);                 // 15I - KV@U1
    k_mm64<<<BH_, 128, 0, stream>>>(KV, U2, U1, 13.0f, -1.0f);                 // 13I - KV@U2
    k_mm64<<<BH_, 128, 0, stream>>>(Vcur, U1, Vnxt, 0.0f, 0.25f);              // 0.25*V@U
    float* t = Vcur; Vcur = Vnxt; Vnxt = t;
  }
  k_mm64_bf16T<<<BH_, 128, 0, stream>>>(Vcur, K3V, W2T);                       // W2^T bf16
  k_attn<<<dim3(N_/64, BH_), 128, 0, stream>>>(Qb, Kl, W2T, attn);
  k_proj<<<dim3(ROWS_/64, C_/64), 128, 0, stream>>>(attn, proj_w, proj_b, out);
}
